// Gate_43748536877293
// MI455X (gfx1250) — matrix operand order repaired and checked by reading, not compiled, not run on hardware
//
#include <hip/hip_runtime.h>
#include <hip/hip_bf16.h>
#include <math.h>

#define N_EXPERTS 256
#define DIM 7168
#define N_GROUPS 8
#define GROUP_SZ 32
#define TOPK_GROUPS 4
#define TOPK 8
#define ROUTE_SCALE 2.5f
#define ROWS_PER_BLOCK 32

typedef __attribute__((ext_vector_type(16))) __bf16 v16bf;
typedef __attribute__((ext_vector_type(8)))  float  v8f;

// A-matrix fragment (16x32 bf16): lane<16 holds row=lane, K = {0..7} U {16..23};
// lane>=16 holds row=lane-16, K = {8..15} U {24..31}.  `p` is pre-offset by half*8.
static __device__ __forceinline__ v16bf load_frag_a(const float* __restrict__ p) {
    float4 f0 = *(const float4*)(p + 0);
    float4 f1 = *(const float4*)(p + 4);
    float4 f2 = *(const float4*)(p + 16);
    float4 f3 = *(const float4*)(p + 20);
    v16bf r;
    r[0]  = (__bf16)f0.x; r[1]  = (__bf16)f0.y; r[2]  = (__bf16)f0.z; r[3]  = (__bf16)f0.w;
    r[4]  = (__bf16)f1.x; r[5]  = (__bf16)f1.y; r[6]  = (__bf16)f1.z; r[7]  = (__bf16)f1.w;
    r[8]  = (__bf16)f2.x; r[9]  = (__bf16)f2.y; r[10] = (__bf16)f2.z; r[11] = (__bf16)f2.w;
    r[12] = (__bf16)f3.x; r[13] = (__bf16)f3.y; r[14] = (__bf16)f3.z; r[15] = (__bf16)f3.w;
    return r;
}

// B-matrix fragment (32x16 bf16): lane<16 holds col=lane, K = {0..7} U {16..23};
// lane>=16 holds col=lane-16, K = {8..15} U {24..31}.  `p` is pre-offset by half*8.
static __device__ __forceinline__ v16bf load_frag_b(const float* __restrict__ p) {
    float4 f0 = *(const float4*)(p + 0);
    float4 f1 = *(const float4*)(p + 4);
    float4 f2 = *(const float4*)(p + 16);
    float4 f3 = *(const float4*)(p + 20);
    v16bf r;
    r[0]  = (__bf16)f0.x; r[1]  = (__bf16)f0.y; r[2]  = (__bf16)f0.z; r[3]  = (__bf16)f0.w;
    r[4]  = (__bf16)f1.x; r[5]  = (__bf16)f1.y; r[6]  = (__bf16)f1.z; r[7]  = (__bf16)f1.w;
    r[8]  = (__bf16)f2.x; r[9]  = (__bf16)f2.y; r[10] = (__bf16)f2.z; r[11] = (__bf16)f2.w;
    r[12] = (__bf16)f3.x; r[13] = (__bf16)f3.y; r[14] = (__bf16)f3.z; r[15] = (__bf16)f3.w;
    return r;
}

#define WMMA_BF16(A, B, C) \
    __builtin_amdgcn_wmma_f32_16x16x32_bf16(false, (A), false, (B), (short)0, (C), false, false)

__global__ __launch_bounds__(256) void moe_gate_kernel(
        const float* __restrict__ x,       // [n, DIM]
        const float* __restrict__ weight,  // [N_EXPERTS, DIM]
        const float* __restrict__ bias,    // [N_EXPERTS]
        float* __restrict__ out,           // [n*8] weights, then [n*8] indices
        int n) {
    __shared__ float s_scores[ROWS_PER_BLOCK][N_EXPERTS];   // 32 KB
    __shared__ float s_bias[N_EXPERTS];                     // 1 KB
    __shared__ float s_part[ROWS_PER_BLOCK][8];             // 1 KB
    __shared__ float s_rv[ROWS_PER_BLOCK];                  // 128 B

    const int tid  = threadIdx.x;
    const int lane = tid & 31;
    const int wave = tid >> 5;      // 0..7 -> experts [wave*32, wave*32+32)
    const int half = lane >> 4;     // 0/1
    const int lr   = lane & 15;

    // stage bias into LDS (consumed after the softmax barriers)
    s_bias[tid] = bias[tid];

    const int row0 = blockIdx.x * ROWS_PER_BLOCK;
    const int rA0 = min(row0 + lr,      n - 1);
    const int rA1 = min(row0 + 16 + lr, n - 1);

    const float* xp0 = x + (size_t)rA0 * DIM + half * 8;
    const float* xp1 = x + (size_t)rA1 * DIM + half * 8;
    const float* wp0 = weight + (size_t)(wave * 32 + lr)      * DIM + half * 8;
    const float* wp1 = weight + (size_t)(wave * 32 + 16 + lr) * DIM + half * 8;

    v8f acc00 = {}, acc01 = {}, acc10 = {}, acc11 = {};

    for (int kc = 0; kc < DIM; kc += 32) {
        // hint the next K-chunk of the streamed activations
        __builtin_prefetch(xp0 + kc + 32, 0, 0);
        __builtin_prefetch(xp1 + kc + 32, 0, 0);

        v16bf b0 = load_frag_b(wp0 + kc);
        v16bf b1 = load_frag_b(wp1 + kc);
        v16bf a0 = load_frag_a(xp0 + kc);
        acc00 = WMMA_BF16(a0, b0, acc00);
        acc01 = WMMA_BF16(a0, b1, acc01);
        v16bf a1 = load_frag_a(xp1 + kc);
        acc10 = WMMA_BF16(a1, b0, acc10);
        acc11 = WMMA_BF16(a1, b1, acc11);
    }

    // D layout: VGPR r, lanes 0-15 -> M=r, lanes 16-31 -> M=8+r; N = lr
    {
        const int mb = half * 8;
        const int e0 = wave * 32 + lr;
        const int e1 = wave * 32 + 16 + lr;
#pragma unroll
        for (int r = 0; r < 8; ++r) {
            s_scores[mb + r][e0]      = acc00[r];
            s_scores[mb + r][e1]      = acc01[r];
            s_scores[16 + mb + r][e0] = acc10[r];
            s_scores[16 + mb + r][e1] = acc11[r];
        }
    }
    __syncthreads();

    // ---- cooperative softmax over 256 experts per row ----
    // 256 threads = 32 rows x 8 chunks of 32 experts
    const int sr = tid >> 3;          // row 0..31
    const int sc = tid & 7;           // chunk 0..7
    const int sb = sc * 32;

    float m = -__builtin_inff();
#pragma unroll 4
    for (int i = 0; i < 32; ++i) m = fmaxf(m, s_scores[sr][sb + i]);
    s_part[sr][sc] = m;
    __syncthreads();
    if (sc == 0) {
        float mm = s_part[sr][0];
#pragma unroll
        for (int i = 1; i < 8; ++i) mm = fmaxf(mm, s_part[sr][i]);
        s_rv[sr] = mm;
    }
    __syncthreads();
    {
        const float mm = s_rv[sr];
        float sum = 0.f;
#pragma unroll 4
        for (int i = 0; i < 32; ++i) {
            float e = __expf(s_scores[sr][sb + i] - mm);
            s_scores[sr][sb + i] = e;
            sum += e;
        }
        s_part[sr][sc] = sum;
    }
    __syncthreads();
    if (sc == 0) {
        float sum = 0.f;
#pragma unroll
        for (int i = 0; i < 8; ++i) sum += s_part[sr][i];
        s_rv[sr] = 1.f / sum;
    }
    __syncthreads();
    {
        const float inv = s_rv[sr];
#pragma unroll 4
        for (int i = 0; i < 32; ++i) s_scores[sr][sb + i] *= inv;
    }
    __syncthreads();

    // ---- grouped top-k, one thread per row ----
    if (tid < ROWS_PER_BLOCK) {
        const int row = row0 + tid;
        if (row < n) {
            // group score = sum of top-2 of (softmax + bias) within each group of 32
            float gs[N_GROUPS];
#pragma unroll
            for (int g = 0; g < N_GROUPS; ++g) {
                float b1 = -__builtin_inff(), b2 = -__builtin_inff();
                const int gb = g * GROUP_SZ;
                for (int i = 0; i < GROUP_SZ; ++i) {
                    float v = s_scores[tid][gb + i] + s_bias[gb + i];
                    if (v > b1) { b2 = b1; b1 = v; }
                    else if (v > b2) { b2 = v; }
                }
                gs[g] = b1 + b2;
            }
            // top-4 groups
            bool keep[N_GROUPS];
#pragma unroll
            for (int g = 0; g < N_GROUPS; ++g) keep[g] = false;
#pragma unroll
            for (int it = 0; it < TOPK_GROUPS; ++it) {
                int bg = -1; float bv = -__builtin_inff();
#pragma unroll
                for (int g = 0; g < N_GROUPS; ++g) {
                    if (!keep[g] && gs[g] > bv) { bv = gs[g]; bg = g; }
                }
                keep[bg] = true;
            }
            // top-8 experts among kept groups (on biased scores);
            // route weight = original softmax prob * scale
            int   idx8[TOPK];
            float wgt8[TOPK];
#pragma unroll
            for (int it = 0; it < TOPK; ++it) {
                int bi = 0; float bv = -__builtin_inff();
                for (int e = 0; e < N_EXPERTS; ++e) {
                    if (!keep[e >> 5]) continue;
                    float v = s_scores[tid][e] + s_bias[e];
                    if (v > bv) { bv = v; bi = e; }
                }
                idx8[it] = bi;
                wgt8[it] = s_scores[tid][bi] * ROUTE_SCALE;
                s_scores[tid][bi] = -__builtin_inff();   // exclude from later picks
            }
            float* out_w = out;
            float* out_i = out + (size_t)n * TOPK;
#pragma unroll
            for (int j = 0; j < TOPK; ++j) {
                out_w[(size_t)row * TOPK + j] = wgt8[j];
                out_i[(size_t)row * TOPK + j] = (float)idx8[j];
            }
        }
    }
}

extern "C" void kernel_launch(void* const* d_in, const int* in_sizes, int n_in,
                              void* d_out, int out_size, void* d_ws, size_t ws_size,
                              hipStream_t stream) {
    const float* x      = (const float*)d_in[0];
    const float* weight = (const float*)d_in[1];
    const float* bias   = (const float*)d_in[2];
    float* out = (float*)d_out;
    const int n = in_sizes[0] / DIM;
    const int blocks = (n + ROWS_PER_BLOCK - 1) / ROWS_PER_BLOCK;
    moe_gate_kernel<<<blocks, 256, 0, stream>>>(x, weight, bias, out, n);
}
